// SimilarityAttention_30202210025964
// MI455X (gfx1250) — compile-verified, matrix operand order checked by reading, not run on hardware
//
#include <hip/hip_runtime.h>

typedef __attribute__((ext_vector_type(16))) _Float16 v16h;
typedef __attribute__((ext_vector_type(8)))  _Float16 v8h;
typedef __attribute__((ext_vector_type(8)))  float    v8f;
typedef __attribute__((ext_vector_type(4)))  float    f4;

#define INPUT_BITS 2048
#define N_KEYS     65536
#define THRESHOLD  1024.0f

// Hamming distance via WMMA GEMV:
//   dist(m) = Qsum + sum_i key[m][i] * (1 - 2*q[i])
// A = 16 keys x 32 bits (f16), B = (1-2q) chunk broadcast to 16 cols (f16),
// C/D = f32 accumulators. All values exact in f16/f32.
__global__ __launch_bounds__(256) void hamming_wmma_kernel(
    const float* __restrict__ query,
    const float* __restrict__ keys,
    float* __restrict__ out) {
  __shared__ _Float16 sB[INPUT_BITS];   // b = 1 - 2q, exact in f16 (+1/-1)
  __shared__ float    sQsum;

  const int tid = threadIdx.x;
  if (tid == 0) sQsum = 0.0f;
  __syncthreads();

  // One pass over the 8KB query: build b in LDS and accumulate Qsum.
  float local = 0.0f;
  #pragma unroll
  for (int i = tid; i < INPUT_BITS; i += 256) {   // 2048/256 = 8, uniform trip count
    float q = query[i];
    local += q;
    sB[i] = (_Float16)(1.0f - 2.0f * q);
  }
  atomicAdd(&sQsum, local);                        // ds_add_f32
  __syncthreads();
  const float qsum = sQsum;

  const int wave = tid >> 5;        // 0..7
  const int lane = tid & 31;
  const int g    = lane >> 4;       // lane half-group (K-chunk selector)
  const int m    = lane & 15;       // A-matrix row within tile
  const int tileKey = (blockIdx.x * 8 + wave) * 16;
  const float* __restrict__ krow = keys + (size_t)(tileKey + m) * INPUT_BITS;

  v8f acc = {};
  #pragma unroll 2
  for (int k0 = 0; k0 < INPUT_BITS; k0 += 32) {
    // A layout (16-bit A 16x32): lanes 0-15 -> K {k0..k0+7, k0+16..k0+23}
    //                            lanes 16-31 -> K {k0+8..k0+15, k0+24..k0+31}
    const int c0 = k0 + g * 8;
    const int c1 = c0 + 16;

    // prefetch 128 floats (512B, 4 K-steps) ahead per row (speculative; OOB tail is harmless)
    __builtin_prefetch(krow + k0 + 128, 0, 3);

    f4 a0 = *(const f4*)(krow + c0);
    f4 a1 = *(const f4*)(krow + c0 + 4);
    f4 a2 = *(const f4*)(krow + c1);
    f4 a3 = *(const f4*)(krow + c1 + 4);

    v16h A;
    A[0]  = (_Float16)a0[0]; A[1]  = (_Float16)a0[1];
    A[2]  = (_Float16)a0[2]; A[3]  = (_Float16)a0[3];
    A[4]  = (_Float16)a1[0]; A[5]  = (_Float16)a1[1];
    A[6]  = (_Float16)a1[2]; A[7]  = (_Float16)a1[3];
    A[8]  = (_Float16)a2[0]; A[9]  = (_Float16)a2[1];
    A[10] = (_Float16)a2[2]; A[11] = (_Float16)a2[3];
    A[12] = (_Float16)a3[0]; A[13] = (_Float16)a3[1];
    A[14] = (_Float16)a3[2]; A[15] = (_Float16)a3[3];

    // B: identical columns -> per-lane K gather mirrors A's K mapping.
    v8h b0 = *(const v8h*)(sB + c0);
    v8h b1 = *(const v8h*)(sB + c1);
    v16h B;
    #pragma unroll
    for (int i = 0; i < 8; ++i) { B[i] = b0[i]; B[i + 8] = b1[i]; }

    acc = __builtin_amdgcn_wmma_f32_16x16x32_f16(
        /*neg_a=*/false, A, /*neg_b=*/false, B,
        /*c_mod=*/(short)0, acc, /*reuse_a=*/false, /*reuse_b=*/false);
  }

  // D[m][n] identical across n. VGPR r: lanes 0-15 -> M=r, lanes 16-31 -> M=8+r.
  if (m == 0) {  // lanes 0 and 16 hold M = g*8 .. g*8+7
    const int base = tileKey + g * 8;
    f4 w0, w1;
    #pragma unroll
    for (int r = 0; r < 4; ++r) {
      w0[r] = ((qsum + acc[r])     <= THRESHOLD) ? 1.0f : 0.0f;
      w1[r] = ((qsum + acc[r + 4]) <= THRESHOLD) ? 1.0f : 0.0f;
    }
    *(f4*)(out + base)     = w0;
    *(f4*)(out + base + 4) = w1;
  }
}

extern "C" void kernel_launch(void* const* d_in, const int* in_sizes, int n_in,
                              void* d_out, int out_size, void* d_ws, size_t ws_size,
                              hipStream_t stream) {
  const float* query = (const float*)d_in[0];   // [INPUT_BITS]
  const float* keys  = (const float*)d_in[1];   // [N_KEYS, INPUT_BITS]
  float*       out   = (float*)d_out;           // [N_KEYS]

  // 8 waves/block * 16 keys/wave = 128 keys per block
  dim3 grid(N_KEYS / 128), block(256);
  hipLaunchKernelGGL(hamming_wmma_kernel, grid, block, 0, stream, query, keys, out);
}
